// ContactHead_2680059592902
// MI455X (gfx1250) — matrix operand order repaired and checked by reading, not compiled, not run on hardware
//
#include <hip/hip_runtime.h>
#include <hip/hip_bf16.h>
#include <math.h>

typedef __attribute__((ext_vector_type(16))) _Float16 v16h;
typedef __attribute__((ext_vector_type(8)))  _Float16 v8h;
typedef __attribute__((ext_vector_type(4)))  _Float16 v4h;
typedef __attribute__((ext_vector_type(8)))  float    v8f;

constexpr int kB   = 32;
constexpr int kC   = 1280;
constexpr int kH   = 32;
constexpr int kW   = 32;
constexpr int kN   = 6890;
constexpr int kHID = 256;
constexpr int kMID = 128;
constexpr int kHW  = kH * kW;          // 1024
constexpr int ROWS = 32;               // rows (points) per block: 2 row-tiles
constexpr int NK1  = kC / 32;          // 40 K-chunks for GEMM1
constexpr int NBLK = (kB * kN) / ROWS; // 6890 exactly

// ---------------------------------------------------------------------------
// Prep: convert weights to f16 and transpose so each lane's WMMA B-fragment
// (16 contiguous halves along K) is one 32-byte global load.
//   wT  [HID][C]   <- reduce_w [C][HID]
//   w1T [MID][HID] <- cls_w1   [HID][MID]
// ---------------------------------------------------------------------------
__global__ void prep_weights_kernel(const float* __restrict__ rw,
                                    const float* __restrict__ w1,
                                    _Float16* __restrict__ wT,
                                    _Float16* __restrict__ w1T) {
    int i = blockIdx.x * blockDim.x + threadIdx.x;
    constexpr int T1 = kHID * kC;      // 327680
    constexpr int T2 = kMID * kHID;    // 32768
    if (i < T1) {
        int d = i / kC, c = i % kC;
        wT[i] = (_Float16)rw[c * kHID + d];
    } else if (i < T1 + T2) {
        int j = i - T1;
        int m = j / kHID, d = j % kHID;
        w1T[j] = (_Float16)w1[d * kMID + m];
    }
}

struct SharedBlk {
    alignas(16) _Float16 A[2][ROWS * 32];   // double-buffered A-tile (4 KB)
    alignas(16) _Float16 O1[ROWS * kHID];   // GEMM1 output, f16 (16 KB)
    float     w4[ROWS][4];                  // bilinear weights (mask folded in)
    int       i4[ROWS][4];                  // clamped corner flat indices
    long long foff[ROWS];                   // batch offset into feat_map
    float     red[ROWS];                    // final per-row reduction
};

__device__ __forceinline__ v16h load_a_frag(const _Float16* rowbase, int hf) {
    const v8h* ap = reinterpret_cast<const v8h*>(rowbase);
    v8h alo = ap[2 * hf];     // K {0-7} or {16-23}
    v8h ahi = ap[2 * hf + 1]; // K {8-15} or {24-31}
    return __builtin_shufflevector(alo, ahi,
           0,1,2,3,4,5,6,7,8,9,10,11,12,13,14,15);
}

__global__ __launch_bounds__(256)
void contact_head_kernel(const float* __restrict__ feat,
                         const float* __restrict__ uv,
                         const float* __restrict__ reduce_b,
                         const float* __restrict__ cls_b1,
                         const float* __restrict__ cls_w2,
                         const float* __restrict__ cls_b2,
                         const _Float16* __restrict__ wT,
                         const _Float16* __restrict__ w1T,
                         float* __restrict__ out) {
    __shared__ SharedBlk S;

    const int tid  = threadIdx.x;
    const int lane = tid & 31;      // wave32 lane
    const int wave = tid >> 5;      // 0..7
    const int nl   = lane & 15;     // N-column within tile / A-row
    const int hf   = lane >> 4;     // lane-half: selects K sub-chunk
    const int mrow = hf * 8;        // C/D: rows m = mrow..mrow+7

    // ---- per-row sampling parameters (align_corners=True, zeros padding) ----
    if (tid < ROWS) {
        int r = blockIdx.x * ROWS + tid;      // flat row = b*N + n
        int b = r / kN;
        float u  = uv[2 * r + 0];
        float vv = uv[2 * r + 1];
        float x = (u  + 1.f) * 0.5f * (kW - 1);
        float y = (vv + 1.f) * 0.5f * (kH - 1);
        float x0f = floorf(x), y0f = floorf(y);
        float wx = x - x0f, wy = y - y0f;
        int x0 = (int)x0f, y0 = (int)y0f, x1 = x0 + 1, y1 = y0 + 1;
        float mx0 = (x0 >= 0 && x0 < kW) ? 1.f : 0.f;
        float mx1 = (x1 >= 0 && x1 < kW) ? 1.f : 0.f;
        float my0 = (y0 >= 0 && y0 < kH) ? 1.f : 0.f;
        float my1 = (y1 >= 0 && y1 < kH) ? 1.f : 0.f;
        int cx0 = x0 < 0 ? 0 : (x0 > kW - 1 ? kW - 1 : x0);
        int cx1 = x1 < 0 ? 0 : (x1 > kW - 1 ? kW - 1 : x1);
        int cy0 = y0 < 0 ? 0 : (y0 > kH - 1 ? kH - 1 : y0);
        int cy1 = y1 < 0 ? 0 : (y1 > kH - 1 ? kH - 1 : y1);
        S.w4[tid][0] = (1.f - wy) * (1.f - wx) * my0 * mx0;
        S.w4[tid][1] = (1.f - wy) * wx         * my0 * mx1;
        S.w4[tid][2] = wy         * (1.f - wx) * my1 * mx0;
        S.w4[tid][3] = wy         * wx         * my1 * mx1;
        S.i4[tid][0] = cy0 * kW + cx0;
        S.i4[tid][1] = cy0 * kW + cx1;
        S.i4[tid][2] = cy1 * kW + cx0;
        S.i4[tid][3] = cy1 * kW + cx1;
        S.foff[tid]  = (long long)b * kC * kHW;
        S.red[tid]   = 0.f;
    }
    __syncthreads();

    // ---- gather role: thread owns (row, 4 channels) per K-chunk ----
    const int grow  = tid >> 3;         // 0..31
    const int kbase = (tid & 7) * 4;    // 0,4,..,28
    const float w00 = S.w4[grow][0], w01 = S.w4[grow][1];
    const float w10 = S.w4[grow][2], w11 = S.w4[grow][3];
    const int   i00 = S.i4[grow][0], i01 = S.i4[grow][1];
    const int   i10 = S.i4[grow][2], i11 = S.i4[grow][3];
    const float* fb = feat + S.foff[grow] + (size_t)kbase * kHW;

    // ---- GEMM1: (32 x 1280) x (1280 x 256) ----
    // wave owns n-tiles {2w, 2w+1} for BOTH 16-row tiles -> 4 accumulators.
    const int nt0 = 2 * wave, nt1 = 2 * wave + 1;
    const _Float16* bb0 = wT + (size_t)(nt0 * 16 + nl) * kC + hf * 16;
    const _Float16* bb1 = wT + (size_t)(nt1 * 16 + nl) * kC + hf * 16;

    // prologue: gather chunk 0 into buffer 0
    {
        #pragma unroll
        for (int e = 0; e < 4; ++e) {
            // keep loop rolled out explicitly below for vector store
        }
        const float* p = fb;
        float g0 = p[i00]*w00 + p[i01]*w01 + p[i10]*w10 + p[i11]*w11; p += kHW;
        float g1 = p[i00]*w00 + p[i01]*w01 + p[i10]*w10 + p[i11]*w11; p += kHW;
        float g2 = p[i00]*w00 + p[i01]*w01 + p[i10]*w10 + p[i11]*w11; p += kHW;
        float g3 = p[i00]*w00 + p[i01]*w01 + p[i10]*w10 + p[i11]*w11;
        v4h pk = { (_Float16)g0, (_Float16)g1, (_Float16)g2, (_Float16)g3 };
        *reinterpret_cast<v4h*>(&S.A[0][grow * 32 + kbase]) = pk;
    }
    __syncthreads();

    v8f acc00 = {}, acc01 = {}, acc10 = {}, acc11 = {};
    for (int kc = 0; kc < NK1; ++kc) {
        const int cur = kc & 1;
        // overlap: gather next chunk into the other buffer while computing
        if (kc + 1 < NK1) {
            const float* p = fb + (size_t)(kc + 1) * 32 * kHW;
            float g0 = p[i00]*w00 + p[i01]*w01 + p[i10]*w10 + p[i11]*w11; p += kHW;
            float g1 = p[i00]*w00 + p[i01]*w01 + p[i10]*w10 + p[i11]*w11; p += kHW;
            float g2 = p[i00]*w00 + p[i01]*w01 + p[i10]*w10 + p[i11]*w11; p += kHW;
            float g3 = p[i00]*w00 + p[i01]*w01 + p[i10]*w10 + p[i11]*w11;
            v4h pk = { (_Float16)g0, (_Float16)g1, (_Float16)g2, (_Float16)g3 };
            *reinterpret_cast<v4h*>(&S.A[cur ^ 1][grow * 32 + kbase]) = pk;
        }

        // A fragments for both row-tiles (shared across this wave's 2 n-tiles)
        v16h a0 = load_a_frag(&S.A[cur][(0  + nl) * 32], hf);
        v16h a1 = load_a_frag(&S.A[cur][(16 + nl) * 32], hf);
        // B fragments (reused by both row-tiles): 32B contiguous per lane
        v16h b0 = *reinterpret_cast<const v16h*>(bb0 + kc * 32);
        v16h b1 = *reinterpret_cast<const v16h*>(bb1 + kc * 32);

        acc00 = __builtin_amdgcn_wmma_f32_16x16x32_f16(false, a0, false, b0,
                                                       (short)0, acc00, false, false);
        acc01 = __builtin_amdgcn_wmma_f32_16x16x32_f16(false, a0, false, b1,
                                                       (short)0, acc01, false, false);
        acc10 = __builtin_amdgcn_wmma_f32_16x16x32_f16(false, a1, false, b0,
                                                       (short)0, acc10, false, false);
        acc11 = __builtin_amdgcn_wmma_f32_16x16x32_f16(false, a1, false, b1,
                                                       (short)0, acc11, false, false);
        __syncthreads();
    }

    // ---- bias + write Out1 tile to LDS as f16 (32 x 256) ----
    {
        float bias0 = reduce_b[nt0 * 16 + nl];
        float bias1 = reduce_b[nt1 * 16 + nl];
        #pragma unroll
        for (int v = 0; v < 8; ++v) {
            S.O1[(mrow + v) * kHID + nt0 * 16 + nl]      = (_Float16)(acc00[v] + bias0);
            S.O1[(mrow + v) * kHID + nt1 * 16 + nl]      = (_Float16)(acc01[v] + bias1);
            S.O1[(16 + mrow + v) * kHID + nt0 * 16 + nl] = (_Float16)(acc10[v] + bias0);
            S.O1[(16 + mrow + v) * kHID + nt1 * 16 + nl] = (_Float16)(acc11[v] + bias1);
        }
    }
    __syncthreads();

    // ---- GEMM2: (32 x 256) x (256 x 128), wave owns MID n-tile `wave` ----
    v8f acc2a = {}, acc2b = {};   // row-tiles 0 and 1
    const _Float16* bb2 = w1T + (size_t)(wave * 16 + nl) * kHID + hf * 16;
    #pragma unroll
    for (int kc = 0; kc < kHID / 32; ++kc) {
        v16h a0 = load_a_frag(&S.O1[(0  + nl) * kHID + kc * 32], hf);
        v16h a1 = load_a_frag(&S.O1[(16 + nl) * kHID + kc * 32], hf);
        v16h b  = *reinterpret_cast<const v16h*>(bb2 + kc * 32);
        acc2a = __builtin_amdgcn_wmma_f32_16x16x32_f16(false, a0, false, b,
                                                       (short)0, acc2a, false, false);
        acc2b = __builtin_amdgcn_wmma_f32_16x16x32_f16(false, a1, false, b,
                                                       (short)0, acc2b, false, false);
    }

    // ---- +b1, ReLU, x cls_w2, reduce over the 128 MID columns ----
    {
        float b1v = cls_b1[wave * 16 + nl];
        float w2v = cls_w2[wave * 16 + nl];
        float pa[8], pb[8];
        #pragma unroll
        for (int v = 0; v < 8; ++v) {
            float ha = acc2a[v] + b1v; ha = ha > 0.f ? ha : 0.f;
            float hb = acc2b[v] + b1v; hb = hb > 0.f ? hb : 0.f;
            pa[v] = ha * w2v;
            pb[v] = hb * w2v;
        }
        // butterfly across the 16 lanes of each half (rows identical per half)
        #pragma unroll
        for (int off = 1; off < 16; off <<= 1) {
            #pragma unroll
            for (int v = 0; v < 8; ++v) {
                pa[v] += __shfl_xor(pa[v], off, 32);
                pb[v] += __shfl_xor(pb[v], off, 32);
            }
        }
        if (nl == 0) {
            #pragma unroll
            for (int v = 0; v < 8; ++v) {
                atomicAdd(&S.red[mrow + v],      pa[v]);
                atomicAdd(&S.red[16 + mrow + v], pb[v]);
            }
        }
    }
    __syncthreads();

    // ---- sigmoid + store ----
    if (tid < ROWS) {
        float logit = S.red[tid] + cls_b2[0];
        out[blockIdx.x * ROWS + tid] = 1.f / (1.f + expf(-logit));
    }
}

extern "C" void kernel_launch(void* const* d_in, const int* in_sizes, int n_in,
                              void* d_out, int out_size, void* d_ws, size_t ws_size,
                              hipStream_t stream) {
    const float* feat = (const float*)d_in[0];   // (B,C,H,W)
    const float* uv   = (const float*)d_in[1];   // (B,N,2)
    const float* rw   = (const float*)d_in[2];   // (C,HID)
    const float* rb   = (const float*)d_in[3];   // (HID)
    const float* w1   = (const float*)d_in[4];   // (HID,MID)
    const float* b1   = (const float*)d_in[5];   // (MID)
    const float* w2   = (const float*)d_in[6];   // (MID)
    const float* b2   = (const float*)d_in[7];   // (1)
    float* out = (float*)d_out;                  // (B,N) f32

    _Float16* wT  = (_Float16*)d_ws;             // [HID][C]   655,360 B
    _Float16* w1T = wT + (size_t)kHID * kC;      // [MID][HID]  65,536 B

    int prepN = kHID * kC + kMID * kHID;         // 360,448
    prep_weights_kernel<<<(prepN + 255) / 256, 256, 0, stream>>>(rw, w1, wT, w1T);
    contact_head_kernel<<<NBLK, 256, 0, stream>>>(feat, uv, rb, b1, w2, b2,
                                                  wT, w1T, out);
}
